// ClusterAttention_7189775254211
// MI455X (gfx1250) — matrix operand order repaired and checked by reading, not compiled, not run on hardware
//
#include <hip/hip_runtime.h>

// ClusterAttention for MI455X (gfx1250): all GEMMs via v_wmma_f32_16x16x32_f16,
// f32 softmax/bias path, one workgroup (8 wave32) per batch element z.
// Feat tile staged via GLOBAL_LOAD_ASYNC_TO_LDS_B128 when the builtin exists.

#define ZB   4096
#define MT   64      // tokens
#define CD   96      // channels
#define NH   3       // heads
#define HD   32      // head dim
#define NQ   288     // 3*C
#define QSCALE 0.17677669529663687f  // 32^-0.5

#if __has_builtin(__builtin_amdgcn_global_load_async_to_lds_b128) && \
    __has_builtin(__builtin_amdgcn_s_wait_asynccnt)
#define USE_ASYNC_LDS 1
#else
#define USE_ASYNC_LDS 0
#endif

typedef _Float16 v16h __attribute__((ext_vector_type(16)));
typedef _Float16 v8h  __attribute__((ext_vector_type(8)));
typedef _Float16 v4h  __attribute__((ext_vector_type(4)));
typedef float    v8f  __attribute__((ext_vector_type(8)));
typedef int      b128t __attribute__((vector_size(16)));   // async b128 payload type

// A-matrix 16x32 f16 fragment (ISA 7.12.2): lane lr holds row M=lr;
// lane-half 0 -> K {kb+0..7, kb+16..23}, lane-half 1 -> K {kb+8..15, kb+24..31}.
__device__ __forceinline__ v16h load_fragA(const _Float16* __restrict__ p, int stride,
                                           int rowBase, int kBase, int lr, int lh) {
  const _Float16* q = p + (rowBase + lr) * stride + kBase + (lh << 3);
  v8h lo = *reinterpret_cast<const v8h*>(q);
  v8h hi = *reinterpret_cast<const v8h*>(q + 16);
  return __builtin_shufflevector(lo, hi, 0,1,2,3,4,5,6,7,8,9,10,11,12,13,14,15);
}

// B-matrix 32x16 f16 fragment: lane lr holds column N=lr; lane-half 0 -> K {kb+0..7, kb+16..23},
// lane-half 1 -> K {kb+8..15, kb+24..31} (same K order as fragA). Source array is
// row-major [N][K] (i.e. the weight for out = A @ W^T, or K rows, or V^T).
__device__ __forceinline__ v16h load_fragB(const _Float16* __restrict__ p, int stride,
                                           int colBase, int kBase, int lr, int lh) {
  const _Float16* q = p + (colBase + lr) * stride + kBase + (lh << 3);
  v8h lo = *reinterpret_cast<const v8h*>(q);
  v8h hi = *reinterpret_cast<const v8h*>(q + 16);
  return __builtin_shufflevector(lo, hi, 0,1,2,3,4,5,6,7,8,9,10,11,12,13,14,15);
}

// ---- kernel 1: zero the pos-max accumulators -------------------------------
__global__ void init_ws_kernel(unsigned int* maxbits) {
  if (threadIdx.x < 2) maxbits[threadIdx.x] = 0u;
}

// ---- kernel 2: global per-axis max of pos (values are uniform[0,1) >= 0) ---
__global__ __launch_bounds__(256) void pos_max_kernel(const float* __restrict__ pos,
                                                      unsigned int* maxbits) {
  __shared__ float r0[256];
  __shared__ float r1[256];
  float m0 = 0.f, m1 = 0.f;
  for (int i = blockIdx.x * 256 + threadIdx.x; i < ZB * MT; i += gridDim.x * 256) {
    m0 = fmaxf(m0, pos[(size_t)i * 2 + 0]);
    m1 = fmaxf(m1, pos[(size_t)i * 2 + 1]);
  }
  r0[threadIdx.x] = m0; r1[threadIdx.x] = m1;
  __syncthreads();
  for (int s = 128; s > 0; s >>= 1) {
    if (threadIdx.x < (unsigned)s) {
      r0[threadIdx.x] = fmaxf(r0[threadIdx.x], r0[threadIdx.x + s]);
      r1[threadIdx.x] = fmaxf(r1[threadIdx.x], r1[threadIdx.x + s]);
    }
    __syncthreads();
  }
  if (threadIdx.x == 0) {
    // non-negative floats compare identically as unsigned int bit patterns
    atomicMax(&maxbits[0], __float_as_uint(r0[0]));
    atomicMax(&maxbits[1], __float_as_uint(r1[0]));
  }
}

// ---- kernel 3: one-time fp32 -> f16 weight conversion into workspace -------
__global__ __launch_bounds__(256) void cvt_weights_kernel(const float* __restrict__ qkv_w,
                                                          const float* __restrict__ proj_w,
                                                          _Float16* __restrict__ wq,
                                                          _Float16* __restrict__ wp) {
  const int i = blockIdx.x * blockDim.x + threadIdx.x;
  if (i < NQ * CD) wq[i] = (_Float16)qkv_w[i];
  if (i < CD * CD) wp[i] = (_Float16)proj_w[i];
}

// ---- kernel 4: fused attention, one block per z ----------------------------
__global__ __launch_bounds__(256) void cluster_attn_kernel(
    const float* __restrict__ pos, const float* __restrict__ feat,
    const int* __restrict__ mask, const float* __restrict__ qkv_b,
    const float* __restrict__ pos_w, const float* __restrict__ pos_b,
    const float* __restrict__ proj_b,
    const unsigned int* __restrict__ maxbits,
    const _Float16* __restrict__ wq, const _Float16* __restrict__ wp,
    float* __restrict__ out)
{
  __shared__ __align__(16) _Float16 sFeat[MT * CD];   // 12 KB  feat tile (f16)
  __shared__ __align__(16) _Float16 sQ[MT * CD];      // 12 KB  Q * scale, [m][h*32+ch]
  __shared__ __align__(16) _Float16 sK[MT * CD];      // 12 KB  K,         [n][h*32+ch]
  __shared__ __align__(16) _Float16 sVt[CD * MT];     // 12 KB  V^T,       [h*32+ch][n]
  // stage buffer (async DMA landing zone) overlays the attn/prob buffers:
  // stage is dead before the per-head loop touches attn/p.
  __shared__ __align__(16) union SU {
    float stage[MT * CD];                 // 24 KB  raw fp32 feat tile
    struct { float attn[MT * MT]; _Float16 p[MT * MT]; } a;  // 16 KB + 8 KB
  } uAB;
  __shared__ __align__(16) _Float16 sO[MT * CD];      // 12 KB  attn output pre-proj
  __shared__ __align__(16) float    sT[NH * MT];      // rank-1 positional term per token
  __shared__ __align__(16) float    sMaskAdd[MT];     // 0 or -100 per key
  __shared__ __align__(16) float    sRed[MT * 4];     // softmax partial reductions

  float*    const sAttn = uAB.a.attn;
  _Float16* const sP    = uAB.a.p;

  const int tid   = threadIdx.x;
  const int z     = blockIdx.x;
  const int lane  = tid & 31;
  const int waveU = __builtin_amdgcn_readfirstlane(tid >> 5);  // wave id, provably uniform
  const int lr    = lane & 15;
  const int lh    = lane >> 4;     // lane-half

  const float* featZ = feat + (size_t)z * (MT * CD);

#if USE_ASYNC_LDS
  // ---- async DMA of the fp32 feat tile straight into LDS (ASYNCcnt path) ----
  {
    auto gsrc = (__attribute__((address_space(1))) b128t*)featZ;
    auto ldst = (__attribute__((address_space(3))) b128t*)uAB.stage;
#pragma unroll
    for (int it = 0; it < (MT * CD * 4) / (256 * 16); ++it) {  // 6 x b128 per thread
      const int v = it * 256 + tid;   // 16-byte vector index
      __builtin_amdgcn_global_load_async_to_lds_b128(gsrc + v, ldst + v, 0, 0);
    }
    __builtin_amdgcn_s_wait_asynccnt(0);
    __syncthreads();
    for (int i4 = tid; i4 < (MT * CD) / 4; i4 += 256) {
      const float4 f = reinterpret_cast<const float4*>(uAB.stage)[i4];
      v4h h4 = { (_Float16)f.x, (_Float16)f.y, (_Float16)f.z, (_Float16)f.w };
      *reinterpret_cast<v4h*>(&sFeat[i4 * 4]) = h4;
    }
  }
#else
  // ---- fallback: vectorized global load + f16 LDS store ----
  for (int i4 = tid; i4 < (MT * CD) / 4; i4 += 256) {
    const float4 f = reinterpret_cast<const float4*>(featZ)[i4];
    v4h h4 = { (_Float16)f.x, (_Float16)f.y, (_Float16)f.z, (_Float16)f.w };
    *reinterpret_cast<v4h*>(&sFeat[i4 * 4]) = h4;
  }
#endif

  // ---- rank-1 positional bias precompute: t[h][i] = posn(i) . pos_w[h] ----
  if (tid < MT) {
    const float inv0 = 1.f / __uint_as_float(maxbits[0]);
    const float inv1 = 1.f / __uint_as_float(maxbits[1]);
    const float p0 = pos[((size_t)z * MT + tid) * 2 + 0] * inv0;
    const float p1 = pos[((size_t)z * MT + tid) * 2 + 1] * inv1;
#pragma unroll
    for (int h = 0; h < NH; ++h)
      sT[h * MT + tid] = p0 * pos_w[h * 2 + 0] + p1 * pos_w[h * 2 + 1];
    sMaskAdd[tid] = (mask[(size_t)z * MT + tid] != 0) ? 0.f : -100.f;
  }
  __syncthreads();

  // ---- QKV GEMM: [64x96] @ [288x96]^T, 4x18 tiles, K = 3x32 ----
  for (int t = waveU; t < 72; t += 8) {
    const int tU = __builtin_amdgcn_readfirstlane(t);
    const int mt = tU & 3, nt = tU >> 2;      // scalar tile indices
    const int mBase = mt * 16, nBase = nt * 16;
    v8f acc = {};
#pragma unroll
    for (int ks = 0; ks < 3; ++ks) {
      v16h a = load_fragA(sFeat, CD, mBase, ks * 32, lr, lh);
      v16h b = load_fragB(wq,    CD, nBase, ks * 32, lr, lh);
      acc = __builtin_amdgcn_wmma_f32_16x16x32_f16(false, a, false, b, (short)0, acc, false, false);
    }
    // head/part are tile-uniform: nt in [0,18), 6 tiles per head, 2 per q/k/v part
    const int hU  = nt / 6;                   // scalar head
    const int sel = (nt % 6) >> 1;            // scalar: 0=q 1=k 2=v
    const int cl  = hU * 32 + ((nt & 1) << 4) + lr;   // packed head-major channel
    const float bv = qkv_b[nBase + lr];
    if (sel == 0) {
#pragma unroll
      for (int r = 0; r < 8; ++r) {
        const int m = mBase + r + lh * 8;
        sQ[m * CD + cl] = (_Float16)((acc[r] + bv) * QSCALE);
      }
    } else if (sel == 1) {
#pragma unroll
      for (int r = 0; r < 8; ++r) {
        const int m = mBase + r + lh * 8;
        sK[m * CD + cl] = (_Float16)(acc[r] + bv);
      }
    } else {
#pragma unroll
      for (int r = 0; r < 8; ++r) {
        const int m = mBase + r + lh * 8;
        sVt[cl * MT + m] = (_Float16)(acc[r] + bv);   // store V transposed
      }
    }
  }
  __syncthreads();

  // ---- per-head attention ----
#pragma unroll 1
  for (int h = 0; h < NH; ++h) {
    // Q @ K^T : 4x4 tiles, K = 32; 2 tiles per wave
#pragma unroll
    for (int tt = 0; tt < 2; ++tt) {
      const int t  = __builtin_amdgcn_readfirstlane(waveU * 2 + tt);
      const int mt = t & 3, nt = t >> 2;
      v16h a = load_fragA(sQ, CD, mt * 16, h * 32, lr, lh);
      v16h b = load_fragB(sK, CD, nt * 16, h * 32, lr, lh);
      v8f acc = {};
      acc = __builtin_amdgcn_wmma_f32_16x16x32_f16(false, a, false, b, (short)0, acc, false, false);
#pragma unroll
      for (int r = 0; r < 8; ++r)
        sAttn[(mt * 16 + r + lh * 8) * MT + nt * 16 + lr] = acc[r];
    }
    __syncthreads();

    // softmax over keys, with bias t[n]-t[m]+pos_b[h] and mask add
    {
      const int row = tid >> 2, qq = tid & 3;   // 4 threads per row, 16 cols each
      const float tRow = sT[h * MT + row];
      const float pb   = pos_b[h];
      float vals[16];
      float vmax = -1e30f;
#pragma unroll
      for (int c0 = 0; c0 < 16; ++c0) {
        const int c   = qq * 16 + c0;
        const float v = sAttn[row * MT + c] + (sT[h * MT + c] - tRow + pb) + sMaskAdd[c];
        vals[c0] = v;
        vmax = fmaxf(vmax, v);
      }
      sRed[row * 4 + qq] = vmax;
      __syncthreads();
      const float m4 = fmaxf(fmaxf(sRed[row * 4 + 0], sRed[row * 4 + 1]),
                             fmaxf(sRed[row * 4 + 2], sRed[row * 4 + 3]));
      float s = 0.f;
#pragma unroll
      for (int c0 = 0; c0 < 16; ++c0) {
        const float e = __expf(vals[c0] - m4);
        vals[c0] = e;
        s += e;
      }
      __syncthreads();
      sRed[row * 4 + qq] = s;
      __syncthreads();
      const float inv = 1.f / (sRed[row * 4 + 0] + sRed[row * 4 + 1] +
                               sRed[row * 4 + 2] + sRed[row * 4 + 3]);
#pragma unroll
      for (int c0 = 0; c0 < 16; ++c0)
        sP[row * MT + qq * 16 + c0] = (_Float16)(vals[c0] * inv);
    }
    __syncthreads();

    // P @ V : 4x2 tiles, K = 2x32; 1 tile per wave
    {
      const int mt = waveU & 3, nt = waveU >> 2;   // scalar; nt 0..1
      v8f acc = {};
#pragma unroll
      for (int ks = 0; ks < 2; ++ks) {
        v16h a = load_fragA(sP,  MT, mt * 16,          ks * 32, lr, lh);
        v16h b = load_fragB(sVt, MT, h * 32 + nt * 16, ks * 32, lr, lh);
        acc = __builtin_amdgcn_wmma_f32_16x16x32_f16(false, a, false, b, (short)0, acc, false, false);
      }
#pragma unroll
      for (int r = 0; r < 8; ++r)
        sO[(mt * 16 + r + lh * 8) * CD + h * 32 + nt * 16 + lr] = (_Float16)acc[r];
    }
    __syncthreads();
  }

  // ---- output projection: [64x96] @ [96x96]^T + b, 4x6 tiles, K = 3x32 ----
  float* outZ = out + (size_t)z * (MT * CD);
  for (int t = waveU; t < 24; t += 8) {
    const int tU = __builtin_amdgcn_readfirstlane(t);
    const int mt = tU & 3, nt = tU >> 2;   // scalar; nt 0..5
    v8f acc = {};
#pragma unroll
    for (int ks = 0; ks < 3; ++ks) {
      v16h a = load_fragA(sO, CD, mt * 16, ks * 32, lr, lh);
      v16h b = load_fragB(wp, CD, nt * 16, ks * 32, lr, lh);
      acc = __builtin_amdgcn_wmma_f32_16x16x32_f16(false, a, false, b, (short)0, acc, false, false);
    }
    const int j    = nt * 16 + lr;
    const float bv = proj_b[j];
#pragma unroll
    for (int r = 0; r < 8; ++r)
      outZ[(mt * 16 + r + lh * 8) * CD + j] = acc[r] + bv;
  }
}

extern "C" void kernel_launch(void* const* d_in, const int* in_sizes, int n_in,
                              void* d_out, int out_size, void* d_ws, size_t ws_size,
                              hipStream_t stream) {
  const float* pos    = (const float*)d_in[0];
  const float* feat   = (const float*)d_in[1];
  const int*   mask   = (const int*)d_in[2];
  const float* qkv_w  = (const float*)d_in[3];
  const float* qkv_b  = (const float*)d_in[4];
  const float* pos_w  = (const float*)d_in[5];
  const float* pos_b  = (const float*)d_in[6];
  const float* proj_w = (const float*)d_in[7];
  const float* proj_b = (const float*)d_in[8];
  float* out = (float*)d_out;

  // workspace layout: [0..7] uint max bits, f16 weights at +64 bytes (16B aligned)
  unsigned int* maxbits = (unsigned int*)d_ws;
  _Float16* wq = (_Float16*)((char*)d_ws + 64);
  _Float16* wp = wq + NQ * CD;

  hipLaunchKernelGGL(init_ws_kernel, dim3(1), dim3(32), 0, stream, maxbits);
  hipLaunchKernelGGL(pos_max_kernel, dim3(64), dim3(256), 0, stream, pos, maxbits);
  hipLaunchKernelGGL(cvt_weights_kernel, dim3((NQ * CD + 255) / 256), dim3(256), 0, stream,
                     qkv_w, proj_w, wq, wp);
  hipLaunchKernelGGL(cluster_attn_kernel, dim3(ZB), dim3(256), 0, stream,
                     pos, feat, mask, qkv_b, pos_w, pos_b, proj_b, maxbits, wq, wp, out);
}
